// MLPF_18313740550512
// MI455X (gfx1250) — compile-verified, matrix operand order checked by reading, not run on hardware
//
#include <hip/hip_runtime.h>
#include <hip/hip_bf16.h>

// ---------------------------------------------------------------------------
// MLPF forward for MI455X (gfx1250): all dense layers go through
// V_WMMA_F32_16X16X32_BF16 (bf16 inputs, f32 accumulate). Weights are
// pre-packed per-GEMM into bf16 [N][Kpad] so B staging is a pure wide copy.
// kNN + aggregation are VALU kernels. Everything launched on `stream`.
// ---------------------------------------------------------------------------

typedef __attribute__((ext_vector_type(16))) __bf16 v16bf;
typedef __attribute__((ext_vector_type(8)))  float  v8f;

union Frag16 { v16bf v; unsigned int u[8]; };

__device__ __forceinline__ unsigned short f2bf(float f) {
    unsigned int u = __float_as_uint(f);
    unsigned int r = (u + 0x7FFFu + ((u >> 16) & 1u)) >> 16;   // RNE
    return (unsigned short)r;
}

// Pack W[K x N] (f32 row-major) -> Wp[N x Kp] (bf16, transposed, zero-padded).
__global__ void pack_w_kernel(const float* __restrict__ W,
                              unsigned short* __restrict__ Wp,
                              int Kd, int Nn, int Kp)
{
    int i = blockIdx.x * 256 + threadIdx.x;
    if (i >= Nn * Kp) return;
    int n = i / Kp, k = i - n * Kp;
    float v = (k < Kd) ? W[(size_t)k * Nn + n] : 0.f;
    Wp[(size_t)n * Kp + k] = f2bf(v);
}

// ---------------------------------------------------------------------------
// Generic GEMM: C[M x N] = act(A[M x K] * W + bias (+ C if accumulate))
// W pre-packed bf16 [Nn x Kp]. Block: 256 threads (8 waves), tile 256x64,
// wave tile 32x64 (2 A-frags x 4 B-frags = 8 WMMA / 32-K step, B reused).
// B staging: one b128 global load + one b128 LDS store per thread per step.
// Requires M % 256 == 0 (true: M = 32768). act: 0 = none, 1 = ELU.
// ---------------------------------------------------------------------------
#define GBM 256
#define GBN 64
#define GBK 32

__global__ __launch_bounds__(256) void gemm_bf16_kernel(
    const float* __restrict__ A, const unsigned short* __restrict__ Wp,
    const float* __restrict__ bias, float* __restrict__ C,
    int M, int Kd, int Nn, int Kp, int accumulate, int act)
{
    __shared__ __align__(16) unsigned short As[GBM * GBK];   // [row][k] 16 KB
    __shared__ __align__(16) unsigned short Bs[GBN * GBK];   // [col][k]  4 KB

    const int bm   = blockIdx.x * GBM;
    const int bn   = blockIdx.y * GBN;
    const int tid  = threadIdx.x;
    const int wave = tid >> 5;
    const int lane = tid & 31;
    const int hf   = lane >> 4;        // which half of the wave
    const int l16  = lane & 15;

    v8f acc[2][4];
    const v8f vzero = {0.f, 0.f, 0.f, 0.f, 0.f, 0.f, 0.f, 0.f};
#pragma unroll
    for (int m = 0; m < 2; ++m)
#pragma unroll
        for (int t = 0; t < 4; ++t) acc[m][t] = vzero;

    const int kLast = Kd - 1;
    // B staging map: 256 threads = 64 rows x 4 uint4 quads (no loop)
    const int bn_row  = tid >> 2;          // 0..63
    const int bn_quad = tid & 3;           // 0..3
    const int gnB     = bn + bn_row;
    const int gnBc    = gnB < Nn ? gnB : (Nn - 1);
    const bool bok    = gnB < Nn;

    for (int k0 = 0; k0 < Kd; k0 += GBK) {
        // ---- stage A tile (256x32) as bf16 pairs, branch-free
        for (int pi = tid; pi < GBM * 16; pi += 256) {
            int r  = pi >> 4;
            int kp = (pi & 15) << 1;
            int gr = bm + r;                       // < M (exact tiling)
            const float* Ar = A + (size_t)gr * Kd;
            int g0 = k0 + kp, g1 = g0 + 1;
            float v0 = Ar[g0 < Kd ? g0 : kLast];
            float v1 = Ar[g1 < Kd ? g1 : kLast];
            v0 = g0 < Kd ? v0 : 0.f;
            v1 = g1 < Kd ? v1 : 0.f;
            ((unsigned int*)As)[pi] =
                (unsigned int)f2bf(v0) | ((unsigned int)f2bf(v1) << 16);
        }
        // ---- stage B tile: pure wide copy from packed bf16 weights
        {
            const uint4* src = (const uint4*)(Wp + (size_t)gnBc * Kp + k0) + bn_quad;
            uint4 v = *src;
            if (!bok) { v.x = 0u; v.y = 0u; v.z = 0u; v.w = 0u; }
            ((uint4*)Bs)[tid] = v;
        }
        __syncthreads();

        // ---- fragments. 16-bit A layout: element pair p covers
        //      K = (p&3)*2 + (p>>2)*16 + hf*8 .. +1  (contiguous 32-bit)
        Frag16 a[2];
#pragma unroll
        for (int m = 0; m < 2; ++m) {
            const int mrow = wave * 32 + m * 16 + l16;
#pragma unroll
            for (int p = 0; p < 8; ++p) {
                int kb = ((p & 3) << 1) + ((p >> 2) << 4) + (hf << 3);
                a[m].u[p] = *(const unsigned int*)&As[mrow * GBK + kb];
            }
        }
#pragma unroll
        for (int t = 0; t < 4; ++t) {
            const int ncol = t * 16 + l16;
            Frag16 b;
#pragma unroll
            for (int p = 0; p < 8; ++p) {
                int kb = ((p & 3) << 1) + ((p >> 2) << 4) + (hf << 3);
                b.u[p] = *(const unsigned int*)&Bs[ncol * GBK + kb];
            }
            acc[0][t] = __builtin_amdgcn_wmma_f32_16x16x32_bf16(
                false, a[0].v, false, b.v, (short)0, acc[0][t], false, false);
            acc[1][t] = __builtin_amdgcn_wmma_f32_16x16x32_bf16(
                false, a[1].v, false, b.v, (short)0, acc[1][t], false, false);
        }
        __syncthreads();
    }

    // ---- epilogue. C layout: VGPR r = row (r + 8*hf) of tile, col = l16.
#pragma unroll
    for (int m = 0; m < 2; ++m) {
#pragma unroll
        for (int t = 0; t < 4; ++t) {
            const int gcol = bn + t * 16 + l16;
            if (gcol >= Nn) continue;
            const float bv = bias ? bias[gcol] : 0.f;
#pragma unroll
            for (int r = 0; r < 8; ++r) {
                const int grow = bm + wave * 32 + m * 16 + r + hf * 8;
                float v = acc[m][t][r] + bv;
                if (accumulate) v += C[(size_t)grow * Nn + gcol];
                if (act == 1) v = v > 0.f ? v : (__expf(v) - 1.f);
                C[(size_t)grow * Nn + gcol] = v;
            }
        }
    }
}

// ---------------------------------------------------------------------------
// kNN top-32 in the learned 4-d space. One thread per query point, candidate
// tiles staged through LDS, register-resident sorted top-K (static indexing).
// Outputs neighbor index (within batch) and weight exp(-10*d2).
// ---------------------------------------------------------------------------
#define KNN 32

__global__ __launch_bounds__(128) void knn_kernel(
    const float* __restrict__ s, int* __restrict__ idx_out,
    float* __restrict__ w_out, int Np)
{
    const int b = blockIdx.y;
    const int i = blockIdx.x * 128 + threadIdx.x;
    const float* sb = s + (size_t)b * Np * 4;
    const float4 si = *(const float4*)&sb[(size_t)i * 4];

    float dist[KNN];
    int   nidx[KNN];
#pragma unroll
    for (int k = 0; k < KNN; ++k) { dist[k] = 1e30f; nidx[k] = 0; }

    __shared__ float4 tile[128];
    for (int j0 = 0; j0 < Np; j0 += 128) {
        __syncthreads();
        tile[threadIdx.x] = *(const float4*)&sb[(size_t)(j0 + threadIdx.x) * 4];
        __syncthreads();
        {   // prefetch next candidate tile (clamped, branch-free)
            int pj = j0 + 128 + (int)threadIdx.x;
            pj = pj < Np ? pj : Np - 1;
            __builtin_prefetch(&sb[(size_t)pj * 4], 0, 1);
        }
        for (int jj = 0; jj < 128; ++jj) {
            float4 sj = tile[jj];
            float dx = si.x - sj.x, dy = si.y - sj.y;
            float dz = si.z - sj.z, dw = si.w - sj.w;
            float d2 = dx * dx + dy * dy + dz * dz + dw * dw;
            if (d2 < dist[KNN - 1]) {          // rare: insert (sorted ascending)
                float dcur = d2; int icur = j0 + jj;
#pragma unroll
                for (int k = 0; k < KNN; ++k) {
                    if (dcur < dist[k]) {
                        float td = dist[k]; int ti = nidx[k];
                        dist[k] = dcur; nidx[k] = icur;
                        dcur = td; icur = ti;
                    }
                }
            }
        }
    }
    const size_t base = ((size_t)b * Np + i) * KNN;
#pragma unroll
    for (int k = 0; k < KNN; ++k) {
        idx_out[base + k] = nidx[k];
        w_out[base + k]   = __expf(-10.f * dist[k]);
    }
}

// ---------------------------------------------------------------------------
// Aggregate: msg[k][p] = h[idx[k]][p] * w[k]; out = [mean_k, max_k] (2P=64).
// One wave per point, lane = property (P = 32).
// ---------------------------------------------------------------------------
__global__ __launch_bounds__(128) void aggregate_kernel(
    const float* __restrict__ h, const int* __restrict__ idx,
    const float* __restrict__ w, float* __restrict__ agg, int Np)
{
    const int lane = threadIdx.x & 31;
    const int i = blockIdx.x * 4 + (threadIdx.x >> 5);   // global point id
    const int b = i / Np;
    const float* hb = h + (size_t)b * Np * 32;
    const size_t base = (size_t)i * KNN;
    float sum = 0.f, mx = -1e30f;
    for (int k = 0; k < KNN; ++k) {
        int   j  = idx[base + k];
        float wk = w[base + k];
        float m  = hb[(size_t)j * 32 + lane] * wk;
        sum += m;
        mx = fmaxf(mx, m);
    }
    agg[(size_t)i * 64 + lane]      = sum * (1.f / KNN);
    agg[(size_t)i * 64 + 32 + lane] = mx;
}

// ---------------------------------------------------------------------------
// LayerNorm over last dim (optionally with residual: out = LN(x + res)).
// One wave per row, wave32 shuffle reduction.
// ---------------------------------------------------------------------------
__global__ __launch_bounds__(256) void ln_kernel(
    const float* __restrict__ x, const float* __restrict__ res,
    float* __restrict__ out, int Wd)
{
    const int row  = blockIdx.x * 8 + (threadIdx.x >> 5);
    const int lane = threadIdx.x & 31;
    const float* xr = x + (size_t)row * Wd;
    const float* rr = res ? res + (size_t)row * Wd : nullptr;

    float vals[10];                        // ceil(296/32) = 10 max
    const int cnt = (Wd + 31) >> 5;
    float sum = 0.f, sq = 0.f;
    for (int c = 0; c < cnt; ++c) {
        int col = c * 32 + lane;
        float v = 0.f;
        if (col < Wd) { v = xr[col]; if (rr) v += rr[col]; }
        vals[c] = v;
        sum += v; sq += v * v;
    }
#pragma unroll
    for (int off = 16; off > 0; off >>= 1) {
        sum += __shfl_xor(sum, off, 32);
        sq  += __shfl_xor(sq,  off, 32);
    }
    const float mean = sum / Wd;
    const float var  = sq / Wd - mean * mean;
    const float inv  = rsqrtf(var + 1e-5f);
    for (int c = 0; c < cnt; ++c) {
        int col = c * 32 + lane;
        if (col < Wd) out[(size_t)row * Wd + col] = (vals[c] - mean) * inv;
    }
}

// Copy src [M x ws] into dst [M x wd] at column offset coff (concat helper).
__global__ void copy_cols_kernel(const float* __restrict__ src,
                                 float* __restrict__ dst,
                                 int M, int ws, int wd, int coff)
{
    int i = blockIdx.x * 256 + threadIdx.x;
    if (i >= M * ws) return;
    int r = i / ws, c = i - r * ws;
    dst[(size_t)r * wd + coff + c] = src[i];
}

// mom = [pt + x1, eta + x2, phi + x3, en + x4]
__global__ void mom_kernel(const float* __restrict__ pt, const float* __restrict__ eta,
                           const float* __restrict__ phi, const float* __restrict__ en,
                           const float* __restrict__ x, float* __restrict__ out, int M)
{
    int i = blockIdx.x * 256 + threadIdx.x;
    if (i >= M) return;
    const float* xi = x + (size_t)i * 34;
    out[(size_t)i * 4 + 0] = pt[i]  + xi[1];
    out[(size_t)i * 4 + 1] = eta[i] + xi[2];
    out[(size_t)i * 4 + 2] = phi[i] + xi[3];
    out[(size_t)i * 4 + 3] = en[i]  + xi[4];
}

// ---------------------------------------------------------------------------
// Host orchestration.
// Param leaf order: d_in[0] = x, then params flattened with sorted dict keys:
//   conv_id[g]: h.W,h.b,out1.W,out2.W,out2.b,s.W,s.b  (base 1 + 7g)
//   conv_reg[g]: same                                 (base 15 + 7g)
//   nn0: (W,b) x4                                     (base 29)
//   nn_charge(37) nn_energy(47) nn_eta(57) nn_id(67) nn_phi(77) nn_pt(87),
//   each: (W,b) x5
// ---------------------------------------------------------------------------
extern "C" void kernel_launch(void* const* d_in, const int* in_sizes, int n_in,
                              void* d_out, int out_size, void* d_ws, size_t ws_size,
                              hipStream_t stream)
{
    const int B = 8, Np = 4096, M = B * Np;
    auto P = [&](int i) -> const float* { return (const float*)d_in[i]; };
    const float* x = P(0);

    // bump allocator over d_ws (f32 elements, 16B aligned)
    float* wsb = (float*)d_ws;
    size_t off = 0;
    auto alloc = [&](size_t elems) -> float* {
        float* p = wsb + off;
        off += (elems + 3) & ~(size_t)3;
        return p;
    };
    float* tA      = alloc((size_t)M * 128);
    float* tB      = alloc((size_t)M * 128);
    float* emb     = alloc((size_t)M * 128);
    float* eid1    = alloc((size_t)M * 128);
    float* eid2    = alloc((size_t)M * 128);
    float* ereg1   = alloc((size_t)M * 128);
    float* ereg2   = alloc((size_t)M * 128);
    float* sbuf    = alloc((size_t)M * 4);
    float* hbuf    = alloc((size_t)M * 32);
    float* wbuf    = alloc((size_t)M * 32);
    int*   idxbuf  = (int*)alloc((size_t)M * 32);
    float* aggbuf  = alloc((size_t)M * 64);
    float* convbuf = alloc((size_t)M * 128);
    float* emb_id  = alloc((size_t)M * 290);
    float* emb_reg = alloc((size_t)M * 296);
    float* predsid = alloc((size_t)M * 6);
    float* t_pt    = alloc((size_t)M);
    float* t_eta   = alloc((size_t)M);
    float* t_phi   = alloc((size_t)M);
    float* t_en    = alloc((size_t)M);
    unsigned short* wpack = (unsigned short*)alloc(24576);  // 49152 bf16 >= 126*320
    if (off * sizeof(float) > ws_size) return;   // insufficient scratch

    auto gemm = [&](const float* A, const float* W, const float* bias, float* C,
                    int m, int k, int n, int accum, int act) {
        const int Kp  = (k + 31) & ~31;
        const int tot = n * Kp;
        pack_w_kernel<<<dim3((tot + 255) / 256), dim3(256), 0, stream>>>(W, wpack, k, n, Kp);
        dim3 g(m / GBM, (n + GBN - 1) / GBN);
        gemm_bf16_kernel<<<g, dim3(256), 0, stream>>>(A, wpack, bias, C, m, k, n, Kp, accum, act);
    };
    auto lnk = [&](const float* in, const float* res, float* out, int w) {
        ln_kernel<<<dim3(M / 8), dim3(256), 0, stream>>>(in, res, out, w);
    };
    auto copycols = [&](const float* src, float* dst, int wsrc, int wdst, int coff) {
        int total = M * wsrc;
        copy_cols_kernel<<<dim3((total + 255) / 256), dim3(256), 0, stream>>>(
            src, dst, M, wsrc, wdst, coff);
    };

    // ---- nn0: 3x (dense + ELU), final dense -> emb [M x 128]
    gemm(x,  P(29), P(30), tA,  M, 34, 126, 0, 1);
    gemm(tA, P(31), P(32), tB,  M, 126, 126, 0, 1);
    gemm(tB, P(33), P(34), tA,  M, 126, 126, 0, 1);
    gemm(tA, P(35), P(36), emb, M, 126, 128, 0, 0);

    // ---- GravNet layer: LN(cur + out1(cur) + out2(agg(kNN(s(cur)), h(cur))))
    auto gravnet = [&](const float* cur, int gb, float* next) {
        gemm(cur, P(gb + 5), P(gb + 6), sbuf, M, 128, 4,  0, 0);   // s
        gemm(cur, P(gb + 0), P(gb + 1), hbuf, M, 128, 32, 0, 0);   // h
        knn_kernel<<<dim3(Np / 128, B), dim3(128), 0, stream>>>(sbuf, idxbuf, wbuf, Np);
        aggregate_kernel<<<dim3(M / 4), dim3(128), 0, stream>>>(hbuf, idxbuf, wbuf, aggbuf, Np);
        gemm(cur,    P(gb + 2), nullptr,   convbuf, M, 128, 128, 0, 0); // out1 (no bias)
        gemm(aggbuf, P(gb + 3), P(gb + 4), convbuf, M, 64,  128, 1, 0); // += out2
        lnk(convbuf, cur, next, 128);                                   // LN(residual)
    };
    gravnet(emb,   1,  eid1);
    gravnet(eid1,  8,  eid2);
    gravnet(emb,   15, ereg1);
    gravnet(ereg1, 22, ereg2);

    // ---- emb_id = concat(x, eid1, eid2) [M x 290]
    copycols(x,    emb_id, 34,  290, 0);
    copycols(eid1, emb_id, 128, 290, 34);
    copycols(eid2, emb_id, 128, 290, 162);

    // ---- FFN tower: 4x LN(ELU(dense)) + final dense
    auto tower = [&](const float* in, int din, int tb, float* out, int dout) {
        gemm(in, P(tb + 0), P(tb + 1), tA, M, din, 126, 0, 1);
        lnk(tA, nullptr, tB, 126);
        gemm(tB, P(tb + 2), P(tb + 3), tA, M, 126, 126, 0, 1);
        lnk(tA, nullptr, tB, 126);
        gemm(tB, P(tb + 4), P(tb + 5), tA, M, 126, 126, 0, 1);
        lnk(tA, nullptr, tB, 126);
        gemm(tB, P(tb + 6), P(tb + 7), tA, M, 126, 126, 0, 1);
        lnk(tA, nullptr, tB, 126);
        gemm(tB, P(tb + 8), P(tb + 9), out, M, 126, dout, 0, 0);
    };

    tower(emb_id, 290, 67, predsid, 6);                 // nn_id
    copycols(predsid, (float*)d_out, 6, 6, 0);          // output 0: preds_id [M x 6]

    // ---- emb_reg = concat(x, ereg1, ereg2, preds_id) [M x 296]
    copycols(x,       emb_reg, 34,  296, 0);
    copycols(ereg1,   emb_reg, 128, 296, 34);
    copycols(ereg2,   emb_reg, 128, 296, 162);
    copycols(predsid, emb_reg, 6,   296, 290);

    tower(emb_reg, 296, 87, t_pt,  1);   // nn_pt
    tower(emb_reg, 296, 57, t_eta, 1);   // nn_eta
    tower(emb_reg, 296, 77, t_phi, 1);   // nn_phi
    tower(emb_reg, 296, 47, t_en,  1);   // nn_energy
    tower(emb_reg, 296, 37, (float*)d_out + (size_t)M * 10, 3);  // nn_charge -> output 2

    mom_kernel<<<dim3((M + 255) / 256), dim3(256), 0, stream>>>(
        t_pt, t_eta, t_phi, t_en, x, (float*)d_out + (size_t)M * 6, M); // output 1: mom
}
